// RNNWithAttention_41781441856026
// MI455X (gfx1250) — compile-verified, matrix operand order checked by reading, not run on hardware
//
#include <hip/hip_runtime.h>
#include <hip/hip_bf16.h>
#include <math.h>

typedef unsigned short u16;
typedef unsigned int   u32;
typedef __attribute__((ext_vector_type(16))) __bf16 bf16x16;
typedef __attribute__((ext_vector_type(8)))  float  f32x8;

constexpr int Bc = 256, Tc = 512, Vc = 128, Ec = 64, Hc = 256, Gc = 4 * Hc;

// ---------- helpers ----------
__device__ __forceinline__ u16 f2bf(float f) {
  u32 u = __float_as_uint(f);
  u32 r = (u + 0x7FFFu + ((u >> 16) & 1u)) >> 16;   // round-to-nearest-even
  return (u16)r;
}
__device__ __forceinline__ float bf2f(u16 h) { return __uint_as_float(((u32)h) << 16); }
__device__ __forceinline__ float sigf(float x) { return 1.0f / (1.0f + expf(-x)); }

// A or B fragment for v_wmma_f32_16x16x32_bf16.
// A (16x32, row-major, ld = K-stride): lane m (0..15) holds K 0..15 of row m,
// lane m+16 holds K 16..31 — one contiguous 32B load per lane.
// B (32x16) built from W stored row-major (N x K): same addressing with "row" = N.
__device__ __forceinline__ bf16x16 frag_ld(const u16* base, int ld) {
  const int lane = threadIdx.x & 31;
  const int r = lane & 15, half = lane >> 4;
  return *(const bf16x16*)(base + (size_t)r * ld + half * 16);
}
__device__ __forceinline__ f32x8 wmma_bf16(bf16x16 a, bf16x16 b, f32x8 c) {
  return __builtin_amdgcn_wmma_f32_16x16x32_bf16(false, a, false, b, (short)0, c,
                                                 false, false);
}

// ---------- fp32 -> bf16 weight conversion ----------
__global__ void cvt_bf16_kernel(const float* __restrict__ s, u16* __restrict__ d, int n) {
  int i = blockIdx.x * blockDim.x + threadIdx.x;
  if (i < n) d[i] = f2bf(s[i]);
}

// ---------- encoder LSTM: 16 WGs x 16 batch rows, full time loop in-kernel ----------
__global__ __launch_bounds__(512) void encoder_kernel(
    const int* __restrict__ x, const float* __restrict__ emb,
    const u16* __restrict__ Wih, const u16* __restrict__ Whh,
    const float* __restrict__ bih, const float* __restrict__ bhh,
    u16* __restrict__ enc_out, float* __restrict__ h_n, float* __restrict__ c_n) {
  __shared__ __align__(32) u16 h_bf[16 * Hc];   // hidden state, bf16
  __shared__ __align__(32) u16 xtile[16 * Ec];  // embedded tokens for this step

  const int tid = threadIdx.x;
  const int wave = tid >> 5, lane = tid & 31;
  const int n = lane & 15, m0 = (lane >> 4) * 8;
  const int b0 = blockIdx.x * 16;
  const int col = wave * 16 + n;  // hidden column owned by this (wave,lane)

  for (int i = tid; i < 16 * Hc; i += 512) h_bf[i] = 0;
  // combined gate biases for this column (i,f,g,o blocks of the 4H gate vector)
  const float bi = bih[col] + bhh[col];
  const float bf_ = bih[Hc + col] + bhh[Hc + col];
  const float bg = bih[2 * Hc + col] + bhh[2 * Hc + col];
  const float bo = bih[3 * Hc + col] + bhh[3 * Hc + col];
  f32x8 c_frag = {};  // cell state tile, lives in registers for all 512 steps
  __syncthreads();

  for (int t = 0; t < Tc; ++t) {
    // Stop LLVM from hoisting the (loop-invariant) weight-fragment loads out of
    // the time loop: hoisting needs ~320 VGPRs and spills to scratch. Weights
    // are L2-resident (2 MB), so re-reading each step is the cheap path.
    asm volatile("" ::: "memory");

    // gather x_t tile: 16 x 64, bf16
    for (int i = tid; i < 16 * Ec; i += 512) {
      int bb = i >> 6, e = i & 63;
      int tok = x[(size_t)(b0 + bb) * Tc + t];
      xtile[i] = f2bf(emb[(size_t)tok * Ec + e]);
    }
    __syncthreads();

    // gates = [x_t | h] @ [Wih ; Whh]^T  (K = 64 + 256), one 16x16 tile/gate/wave
    f32x8 gi = {}, gf = {}, gg = {}, go = {};
#pragma unroll 1
    for (int k = 0; k < Ec; k += 32) {
      bf16x16 a = frag_ld(xtile + k, Ec);
      bf16x16 w0 = frag_ld(Wih + (size_t)(0 * Hc + wave * 16) * Ec + k, Ec);
      bf16x16 w1 = frag_ld(Wih + (size_t)(1 * Hc + wave * 16) * Ec + k, Ec);
      bf16x16 w2 = frag_ld(Wih + (size_t)(2 * Hc + wave * 16) * Ec + k, Ec);
      bf16x16 w3 = frag_ld(Wih + (size_t)(3 * Hc + wave * 16) * Ec + k, Ec);
      gi = wmma_bf16(a, w0, gi);
      gf = wmma_bf16(a, w1, gf);
      gg = wmma_bf16(a, w2, gg);
      go = wmma_bf16(a, w3, go);
    }
#pragma unroll 1
    for (int k = 0; k < Hc; k += 32) {
      bf16x16 a = frag_ld(h_bf + k, Hc);
      bf16x16 w0 = frag_ld(Whh + (size_t)(0 * Hc + wave * 16) * Hc + k, Hc);
      bf16x16 w1 = frag_ld(Whh + (size_t)(1 * Hc + wave * 16) * Hc + k, Hc);
      bf16x16 w2 = frag_ld(Whh + (size_t)(2 * Hc + wave * 16) * Hc + k, Hc);
      bf16x16 w3 = frag_ld(Whh + (size_t)(3 * Hc + wave * 16) * Hc + k, Hc);
      gi = wmma_bf16(a, w0, gi);
      gf = wmma_bf16(a, w1, gf);
      gg = wmma_bf16(a, w2, gg);
      go = wmma_bf16(a, w3, go);
    }
    __syncthreads();  // all LDS reads of h_bf done before overwrite

    float hn8[8];
    for (int r = 0; r < 8; ++r) {
      float ii = sigf(gi[r] + bi), ff = sigf(gf[r] + bf_);
      float gv = tanhf(gg[r] + bg), oo = sigf(go[r] + bo);
      float cc = ff * c_frag[r] + ii * gv;
      c_frag[r] = cc;
      hn8[r] = oo * tanhf(cc);
    }
    for (int r = 0; r < 8; ++r) {
      u16 hb = f2bf(hn8[r]);
      h_bf[(m0 + r) * Hc + col] = hb;
      enc_out[((size_t)(b0 + m0 + r) * Tc + t) * Hc + col] = hb;  // (B,T,H) bf16
    }
    if (t == Tc - 1) {
      for (int r = 0; r < 8; ++r) {
        h_n[(size_t)(b0 + m0 + r) * Hc + col] = hn8[r];
        c_n[(size_t)(b0 + m0 + r) * Hc + col] = c_frag[r];
      }
    }
    __syncthreads();
  }
}

// ---------- keys_proj = enc_out @ Wk^T : one 16x16 tile per wave ----------
__global__ __launch_bounds__(256) void keys_kernel(const u16* __restrict__ enc,
                                                   const u16* __restrict__ Wk,
                                                   u16* __restrict__ keys) {
  const int wave = threadIdx.x >> 5, lane = threadIdx.x & 31;
  const int tile = blockIdx.x * 8 + wave;
  const int mt = tile >> 4, nt = tile & 15;  // 8192 M-tiles x 16 N-tiles
  const u16* A = enc + (size_t)mt * 16 * Hc;
  f32x8 acc = {};
  for (int k = 0; k < Hc; k += 32)
    acc = wmma_bf16(frag_ld(A + k, Hc), frag_ld(Wk + (size_t)(nt * 16) * Hc + k, Hc), acc);
  const int n = lane & 15, m0 = (lane >> 4) * 8;
  for (int r = 0; r < 8; ++r)
    keys[((size_t)mt * 16 + m0 + r) * Hc + nt * 16 + n] = f2bf(acc[r]);
}

// ---------- decoder: attention + LSTM + fused FC, 16 WGs x 16 batch rows ----------
__global__ __launch_bounds__(512) void decoder_kernel(
    const int* __restrict__ x, const float* __restrict__ emb,
    const float* __restrict__ vvec,
    const u16* __restrict__ Wq, const u16* __restrict__ Wih, const u16* __restrict__ Whh,
    const float* __restrict__ bih, const float* __restrict__ bhh,
    const u16* __restrict__ Wfc, const float* __restrict__ bfc,
    const u16* __restrict__ keys, const u16* __restrict__ enc,
    const float* __restrict__ h0, const float* __restrict__ c0,
    float* __restrict__ logits, float* __restrict__ h_f, float* __restrict__ c_f) {
  constexpr int DI = Ec + Hc;                       // 320: decoder input [emb|context]
  __shared__ __align__(32) u16 h_bf[16 * Hc];       // 8 KB
  __shared__ __align__(32) u16 dec_in[16 * DI];     // 10 KB
  __shared__ __align__(32) u16 q_lds[16 * Hc];      // 8 KB (bf16 query)
  __shared__ float eng[16 * Tc];                    // 32 KB energies -> softmax weights
  __shared__ float v_lds[Hc];                       // 1 KB

  const int tid = threadIdx.x;
  const int wave = tid >> 5, lane = tid & 31;
  const int n = lane & 15, m0 = (lane >> 4) * 8;
  const int b0 = blockIdx.x * 16;
  const int col = wave * 16 + n;

  for (int i = tid; i < 16 * Hc; i += 512)
    h_bf[i] = f2bf(h0[(size_t)(b0 + (i >> 8)) * Hc + (i & 255)]);
  for (int i = tid; i < Hc; i += 512) v_lds[i] = vvec[i];
  f32x8 c_frag;
  for (int r = 0; r < 8; ++r) c_frag[r] = c0[(size_t)(b0 + m0 + r) * Hc + col];
  const float bi = bih[col] + bhh[col];
  const float bf_ = bih[Hc + col] + bhh[Hc + col];
  const float bg = bih[2 * Hc + col] + bhh[2 * Hc + col];
  const float bo = bih[3 * Hc + col] + bhh[3 * Hc + col];
  const float bfcv = (wave < 8) ? bfc[wave * 16 + n] : 0.0f;
  __syncthreads();

  for (int t = 0; t < Tc; ++t) {
    asm volatile("" ::: "memory");  // prevent weight-fragment hoist/spill (see encoder)

    // ---- Phase A: q = h @ Wq^T (one tile per wave) + gather emb_t ----
    {
      f32x8 qa = {};
      for (int k = 0; k < Hc; k += 32)
        qa = wmma_bf16(frag_ld(h_bf + k, Hc),
                       frag_ld(Wq + (size_t)(wave * 16) * Hc + k, Hc), qa);
      for (int r = 0; r < 8; ++r) q_lds[(m0 + r) * Hc + col] = f2bf(qa[r]);
    }
    for (int i = tid; i < 16 * Ec; i += 512) {
      int bb = i >> 6, e = i & 63;
      int tok = x[(size_t)(b0 + bb) * Tc + t];
      dec_in[bb * DI + e] = f2bf(emb[(size_t)tok * Ec + e]);
    }
    __syncthreads();

    // ---- Phase B: energy[b][t2] = v . tanh(q_b + keys[b,t2,:]) (+mask) ----
    {
      const int bb = lane & 15, half = lane >> 4;
      for (int tt = 0; tt < Tc / 16; ++tt) {
        const int t2 = wave * (Tc / 16) + tt;
        const u16* kp = keys + ((size_t)(b0 + bb) * Tc + t2) * Hc + half * 128;
        const u16* qp = q_lds + bb * Hc + half * 128;
        float s = 0.0f;
        for (int j = 0; j < 128; j += 8) {
          uint4 u = *(const uint4*)(kp + j);
          float k0 = __uint_as_float(u.x << 16), k1 = __uint_as_float(u.x & 0xffff0000u);
          float k2 = __uint_as_float(u.y << 16), k3 = __uint_as_float(u.y & 0xffff0000u);
          float k4 = __uint_as_float(u.z << 16), k5 = __uint_as_float(u.z & 0xffff0000u);
          float k6 = __uint_as_float(u.w << 16), k7 = __uint_as_float(u.w & 0xffff0000u);
          const int hh = half * 128 + j;
          s += v_lds[hh + 0] * tanhf(bf2f(qp[j + 0]) + k0);
          s += v_lds[hh + 1] * tanhf(bf2f(qp[j + 1]) + k1);
          s += v_lds[hh + 2] * tanhf(bf2f(qp[j + 2]) + k2);
          s += v_lds[hh + 3] * tanhf(bf2f(qp[j + 3]) + k3);
          s += v_lds[hh + 4] * tanhf(bf2f(qp[j + 4]) + k4);
          s += v_lds[hh + 5] * tanhf(bf2f(qp[j + 5]) + k5);
          s += v_lds[hh + 6] * tanhf(bf2f(qp[j + 6]) + k6);
          s += v_lds[hh + 7] * tanhf(bf2f(qp[j + 7]) + k7);
        }
        s += __shfl_xor(s, 16);
        if (half == 0) {
          int tok = x[(size_t)(b0 + bb) * Tc + t2];
          eng[bb * Tc + t2] = (tok == 0) ? -1e9f : s;
        }
      }
    }
    __syncthreads();

    // ---- Phase C: softmax over t2, one row per wave; weights back into eng ----
    {
      const int bb = wave;
      float mx = -3.4e38f;
      for (int j = lane; j < Tc; j += 32) mx = fmaxf(mx, eng[bb * Tc + j]);
      for (int o = 16; o > 0; o >>= 1) mx = fmaxf(mx, __shfl_xor(mx, o));
      float s = 0.0f;
      for (int j = lane; j < Tc; j += 32) {
        float e = expf(eng[bb * Tc + j] - mx);
        eng[bb * Tc + j] = e;
        s += e;
      }
      for (int o = 16; o > 0; o >>= 1) s += __shfl_xor(s, o);
      const float inv = 1.0f / s;
      for (int j = lane; j < Tc; j += 32) eng[bb * Tc + j] *= inv;
    }
    __syncthreads();

    // ---- Phase D: context[b] = sum_t w[b][t] * enc_out[b][t][:] ----
    {
      const int bb = tid >> 5, hb = (tid & 31) * 8;
      float ctx[8] = {0, 0, 0, 0, 0, 0, 0, 0};
      for (int t2 = 0; t2 < Tc; ++t2) {
        float wt = eng[bb * Tc + t2];
        uint4 u = *(const uint4*)(enc + ((size_t)(b0 + bb) * Tc + t2) * Hc + hb);
        ctx[0] += wt * __uint_as_float(u.x << 16);
        ctx[1] += wt * __uint_as_float(u.x & 0xffff0000u);
        ctx[2] += wt * __uint_as_float(u.y << 16);
        ctx[3] += wt * __uint_as_float(u.y & 0xffff0000u);
        ctx[4] += wt * __uint_as_float(u.z << 16);
        ctx[5] += wt * __uint_as_float(u.z & 0xffff0000u);
        ctx[6] += wt * __uint_as_float(u.w << 16);
        ctx[7] += wt * __uint_as_float(u.w & 0xffff0000u);
      }
      for (int j = 0; j < 8; ++j) dec_in[bb * DI + Ec + hb + j] = f2bf(ctx[j]);
    }
    __syncthreads();

    // ---- Phase E: gates = [dec_in | h] @ [Wih_d ; Whh_d]^T, LSTM update ----
    f32x8 gi = {}, gf = {}, gg = {}, go = {};
#pragma unroll 1
    for (int k = 0; k < DI; k += 32) {
      bf16x16 a = frag_ld(dec_in + k, DI);
      bf16x16 w0 = frag_ld(Wih + (size_t)(0 * Hc + wave * 16) * DI + k, DI);
      bf16x16 w1 = frag_ld(Wih + (size_t)(1 * Hc + wave * 16) * DI + k, DI);
      bf16x16 w2 = frag_ld(Wih + (size_t)(2 * Hc + wave * 16) * DI + k, DI);
      bf16x16 w3 = frag_ld(Wih + (size_t)(3 * Hc + wave * 16) * DI + k, DI);
      gi = wmma_bf16(a, w0, gi);
      gf = wmma_bf16(a, w1, gf);
      gg = wmma_bf16(a, w2, gg);
      go = wmma_bf16(a, w3, go);
    }
#pragma unroll 1
    for (int k = 0; k < Hc; k += 32) {
      bf16x16 a = frag_ld(h_bf + k, Hc);
      bf16x16 w0 = frag_ld(Whh + (size_t)(0 * Hc + wave * 16) * Hc + k, Hc);
      bf16x16 w1 = frag_ld(Whh + (size_t)(1 * Hc + wave * 16) * Hc + k, Hc);
      bf16x16 w2 = frag_ld(Whh + (size_t)(2 * Hc + wave * 16) * Hc + k, Hc);
      bf16x16 w3 = frag_ld(Whh + (size_t)(3 * Hc + wave * 16) * Hc + k, Hc);
      gi = wmma_bf16(a, w0, gi);
      gf = wmma_bf16(a, w1, gf);
      gg = wmma_bf16(a, w2, gg);
      go = wmma_bf16(a, w3, go);
    }
    __syncthreads();  // all reads of h_bf done

    float hn8[8];
    for (int r = 0; r < 8; ++r) {
      float ii = sigf(gi[r] + bi), ff = sigf(gf[r] + bf_);
      float gv = tanhf(gg[r] + bg), oo = sigf(go[r] + bo);
      float cc = ff * c_frag[r] + ii * gv;
      c_frag[r] = cc;
      hn8[r] = oo * tanhf(cc);
    }
    for (int r = 0; r < 8; ++r) h_bf[(m0 + r) * Hc + col] = f2bf(hn8[r]);
    if (t == Tc - 1) {
      for (int r = 0; r < 8; ++r) {
        h_f[(size_t)(b0 + m0 + r) * Hc + col] = hn8[r];
        c_f[(size_t)(b0 + m0 + r) * Hc + col] = c_frag[r];
      }
    }
    __syncthreads();  // h_bf visible

    // ---- Phase F: logits[b,t,:] = h_new @ Wfc^T + bfc (waves 0..7, V=128) ----
    if (wave < 8) {
      f32x8 acc = {};
      for (int k = 0; k < Hc; k += 32)
        acc = wmma_bf16(frag_ld(h_bf + k, Hc),
                        frag_ld(Wfc + (size_t)(wave * 16) * Hc + k, Hc), acc);
      for (int r = 0; r < 8; ++r)
        logits[((size_t)(b0 + m0 + r) * Tc + t) * Vc + wave * 16 + n] = acc[r] + bfcv;
    }
    __syncthreads();
  }
}

// ---------- host launch ----------
extern "C" void kernel_launch(void* const* d_in, const int* in_sizes, int n_in,
                              void* d_out, int out_size, void* d_ws, size_t ws_size,
                              hipStream_t stream) {
  (void)in_sizes; (void)n_in; (void)out_size; (void)ws_size;
  const int*   x     = (const int*)  d_in[0];
  const float* emb   = (const float*)d_in[1];
  const float* Wih_e = (const float*)d_in[2];
  const float* Whh_e = (const float*)d_in[3];
  const float* bih_e = (const float*)d_in[4];
  const float* bhh_e = (const float*)d_in[5];
  const float* Wq    = (const float*)d_in[6];
  const float* Wk    = (const float*)d_in[7];
  const float* vv    = (const float*)d_in[8];
  const float* Wih_d = (const float*)d_in[9];
  const float* Whh_d = (const float*)d_in[10];
  const float* bih_d = (const float*)d_in[11];
  const float* bhh_d = (const float*)d_in[12];
  const float* Wfc   = (const float*)d_in[13];
  const float* bfc   = (const float*)d_in[14];

  char* ws = (char*)d_ws;
  size_t off = 0;
  auto take = [&](size_t bytes) -> char* {
    char* p = ws + off;
    off = (off + bytes + 255) & ~(size_t)255;
    return p;
  };
  u16* wih_e = (u16*)take((size_t)Gc * Ec * 2);
  u16* whh_e = (u16*)take((size_t)Gc * Hc * 2);
  u16* wq    = (u16*)take((size_t)Hc * Hc * 2);
  u16* wk    = (u16*)take((size_t)Hc * Hc * 2);
  u16* wih_d = (u16*)take((size_t)Gc * (Ec + Hc) * 2);
  u16* whh_d = (u16*)take((size_t)Gc * Hc * 2);
  u16* wfc   = (u16*)take((size_t)Vc * Hc * 2);
  u16* encb  = (u16*)take((size_t)Bc * Tc * Hc * 2);
  u16* keysb = (u16*)take((size_t)Bc * Tc * Hc * 2);
  float* h_n = (float*)take((size_t)Bc * Hc * 4);
  float* c_n = (float*)take((size_t)Bc * Hc * 4);

  float* logits = (float*)d_out;
  float* h_f = logits + (size_t)Bc * Tc * Vc;
  float* c_f = h_f + (size_t)Bc * Hc;

  auto cvt = [&](const float* s, u16* d, int nElems) {
    cvt_bf16_kernel<<<(nElems + 255) / 256, 256, 0, stream>>>(s, d, nElems);
  };
  cvt(Wih_e, wih_e, Gc * Ec);
  cvt(Whh_e, whh_e, Gc * Hc);
  cvt(Wq, wq, Hc * Hc);
  cvt(Wk, wk, Hc * Hc);
  cvt(Wih_d, wih_d, Gc * (Ec + Hc));
  cvt(Whh_d, whh_d, Gc * Hc);
  cvt(Wfc, wfc, Vc * Hc);

  encoder_kernel<<<Bc / 16, 512, 0, stream>>>(x, emb, wih_e, whh_e, bih_e, bhh_e,
                                              encb, h_n, c_n);
  keys_kernel<<<(Bc * Tc / 16) * 16 / 8, 256, 0, stream>>>(encb, wk, keysb);
  decoder_kernel<<<Bc / 16, 512, 0, stream>>>(x, emb, vv, wq, wih_d, whh_d, bih_d,
                                              bhh_d, wfc, bfc, keysb, encb, h_n, c_n,
                                              logits, h_f, c_f);
}
